// TokenSelector_83708912599683
// MI455X (gfx1250) — matrix operand order repaired and checked by reading, not compiled, not run on hardware
//
#include <hip/hip_runtime.h>
#include <hip/hip_bf16.h>

// ---------------- problem constants ----------------
#define BB     32          // batch
#define SS     65536       // sequence
#define DD     32          // embed dim
#define HH     16          // hidden
#define KSEL   6553        // top-k per row = int(0.1 * 65536)
#define NBINS  4096        // score histogram bins over (0,1)

typedef __attribute__((ext_vector_type(16))) _Float16 v16h;
typedef __attribute__((ext_vector_type(8)))  float    v8f;

__device__ __forceinline__ int score_bin(float s) {
    int bin = (int)(s * (float)NBINS);
    bin = bin < 0 ? 0 : bin;
    bin = bin > (NBINS - 1) ? (NBINS - 1) : bin;
    return bin;
}

// DPP16 butterfly-add step: adds the value from a permuted lane (VALU only,
// no LDS). CTRL must be a compile-time constant for the builtin, hence the
// template parameter. All permutations stay within each 16-lane DPP row, so
// the reduction is confined to each half of the wave32.
template <int CTRL>
__device__ __forceinline__ float dpp_add(float v) {
    int x = __builtin_amdgcn_update_dpp(0, __float_as_int(v), CTRL,
                                        /*row_mask=*/0xF, /*bank_mask=*/0xF,
                                        /*bound_ctrl=*/true);
    return v + __int_as_float(x);
}

// Sum over the 16 lanes of each wave32 half. After quad_perm xor1/xor2 each
// quad holds a uniform partial sum, so ROW_HALF_MIRROR pairs quads and
// ROW_MIRROR pairs octs; result is uniform across the 16-lane group.
__device__ __forceinline__ float reduce16(float v) {
    v = dpp_add<0xB1>(v);    // quad_perm(1,0,3,2)  : xor 1
    v = dpp_add<0x4E>(v);    // quad_perm(2,3,0,1)  : xor 2
    v = dpp_add<0x141>(v);   // ROW_HALF_MIRROR     : cross-quad
    v = dpp_add<0x140>(v);   // ROW_MIRROR          : cross-oct
    return v;
}

// ---------------- 0: zero the histogram ----------------
__global__ void zero_kernel(unsigned int* __restrict__ p, int n) {
    int i = blockIdx.x * 256 + threadIdx.x;
    if (i < n) p[i] = 0u;
}

// ---------------- 1: fused WMMA scorer + histogram ----------------
// grid: 2048 blocks x 256 threads. Each block = 8 waves x 8 tiles x 16 tokens
// = 1024 tokens; 64 blocks per batch row (b constant per block).
__global__ void __launch_bounds__(256)
score_kernel(const float* __restrict__ emb,
             const float* __restrict__ W1,
             const float* __restrict__ b1,
             const float* __restrict__ W2,
             const float* __restrict__ b2,
             float* __restrict__ scores,
             unsigned int* __restrict__ hist) {
    const int lane = threadIdx.x & 31;
    const int wave = threadIdx.x >> 5;
    const int half = lane >> 4;   // which 16-lane half of the wave32
    const int l16  = lane & 15;

    const int b       = blockIdx.x >> 6;                       // 64 blocks / row
    const int tokBase = (blockIdx.x & 63) * 1024 + wave * 128; // 8 tiles of 16

    __shared__ unsigned int lhist[NBINS];
    for (int i = threadIdx.x; i < NBINS; i += 256) lhist[i] = 0u;
    __syncthreads();

    // B fragment: W1 is 32x16 (K x N), row-major [k*HH + n].
    // B layout used here (contiguous k, same as A): lanes 0-15 hold K=0..15 (N=lane), lanes 16-31 hold K=16..31.
    v16h bfrag;
    {
        const int ko = half * 16;
#pragma unroll
        for (int e = 0; e < 16; ++e)
            bfrag[e] = (_Float16)W1[(ko + e) * HH + l16];
    }
    const float b1n = b1[l16];
    const float w2n = W2[l16];
    const float b2s = b2[0];

    const float* rowBase = emb + (size_t)b * SS * DD;
    const int kb = half * 16;  // A layout: lanes 0-15 K 0-15, lanes 16-31 K 16-31

#pragma unroll 4
    for (int it = 0; it < 8; ++it) {
        const int t0 = tokBase + it * 16;
        const float* rowp = rowBase + (size_t)(t0 + l16) * DD;
        if (it + 1 < 8)
            __builtin_prefetch(rowBase + (size_t)(t0 + 16 + l16) * DD, 0, 1);

        float4 f0 = *(const float4*)(rowp + kb);
        float4 f1 = *(const float4*)(rowp + kb + 4);
        float4 f2 = *(const float4*)(rowp + kb + 8);
        float4 f3 = *(const float4*)(rowp + kb + 12);

        v16h a;
        a[0]  = (_Float16)f0.x; a[1]  = (_Float16)f0.y;
        a[2]  = (_Float16)f0.z; a[3]  = (_Float16)f0.w;
        a[4]  = (_Float16)f1.x; a[5]  = (_Float16)f1.y;
        a[6]  = (_Float16)f1.z; a[7]  = (_Float16)f1.w;
        a[8]  = (_Float16)f2.x; a[9]  = (_Float16)f2.y;
        a[10] = (_Float16)f2.z; a[11] = (_Float16)f2.w;
        a[12] = (_Float16)f3.x; a[13] = (_Float16)f3.y;
        a[14] = (_Float16)f3.z; a[15] = (_Float16)f3.w;

        v8f c = {};
        // h[16x16] = tokens(16x32) x W1(32x16)
        c = __builtin_amdgcn_wmma_f32_16x16x32_f16(
                false, a, false, bfrag, (short)0, c, false, false);

        // Layer 2: score[m] = sigmoid(b2 + sum_n relu(h[m,n]+b1[n]) * W2[n]).
        // C layout: VGPR r -> row m=r (lanes 0-15, n=lane) / m=8+r (lanes 16-31).
        float s[8];
#pragma unroll
        for (int r = 0; r < 8; ++r) {
            float p = fmaxf(c[r] + b1n, 0.0f) * w2n;
            p = reduce16(p);                       // VALU-only DPP reduction
            s[r] = 1.0f / (1.0f + __expf(-(p + b2s)));
        }

        if (l16 == 0) {
            float* sp = scores + (size_t)b * SS + t0 + half * 8;
            *(float4*)(sp)     = make_float4(s[0], s[1], s[2], s[3]);
            *(float4*)(sp + 4) = make_float4(s[4], s[5], s[6], s[7]);
#pragma unroll
            for (int r = 0; r < 8; ++r)
                atomicAdd(&lhist[score_bin(s[r])], 1u);
        }
    }

    __syncthreads();
    for (int i = threadIdx.x; i < NBINS; i += 256) {
        unsigned int v = lhist[i];
        if (v) atomicAdd(&hist[b * NBINS + i], v);
    }
}

// ---------------- 2: find cutoff bin per batch row ----------------
// params[b*4] = {cut_bin, count_above, remaining, 0}
__global__ void threshold_kernel(const unsigned int* __restrict__ hist,
                                 int* __restrict__ params) {
    if (threadIdx.x != 0) return;
    const int b = blockIdx.x;
    int cum = 0, cut = 0, above = 0, remaining = KSEL;
    for (int bin = NBINS - 1; bin >= 0; --bin) {
        int c = (int)hist[b * NBINS + bin];
        if (cum + c >= KSEL) { cut = bin; above = cum; remaining = KSEL - cum; break; }
        cum += c;
    }
    params[b * 4 + 0] = cut;
    params[b * 4 + 1] = above;
    params[b * 4 + 2] = remaining;
    params[b * 4 + 3] = 0;
}

// ---------------- 3: per-block hi/eq counts (256 tokens/block) ----------------
__global__ void __launch_bounds__(256)
count_kernel(const float* __restrict__ scores, const int* __restrict__ params,
             int* __restrict__ blkHi, int* __restrict__ blkEq) {
    const int blk = blockIdx.x;
    const int b   = blk >> 8;                       // 256 blocks per row
    const int t   = (blk & 255) * 256 + threadIdx.x;
    const int cut = params[b * 4 + 0];

    __shared__ int cHi, cEq;
    if (threadIdx.x == 0) { cHi = 0; cEq = 0; }
    __syncthreads();

    const int bin = score_bin(scores[(size_t)b * SS + t]);
    if (bin > cut)       atomicAdd(&cHi, 1);
    else if (bin == cut) atomicAdd(&cEq, 1);
    __syncthreads();

    if (threadIdx.x == 0) { blkHi[blk] = cHi; blkEq[blk] = cEq; }
}

// ---------------- 4: per-row serial scan of block bases (deterministic) ----------------
__global__ void scan_kernel(const int* __restrict__ blkHi, const int* __restrict__ blkEq,
                            int* __restrict__ hiBase, int* __restrict__ eqBase) {
    if (threadIdx.x != 0) return;
    const int b = blockIdx.x;
    int rh = 0, re = 0;
    for (int i = 0; i < 256; ++i) {
        const int blk = b * 256 + i;
        hiBase[blk] = rh; rh += blkHi[blk];
        eqBase[blk] = re; re += blkEq[blk];
    }
}

// ---------------- 5: deterministic compaction + gather ----------------
__global__ void __launch_bounds__(256)
select_kernel(const float* __restrict__ emb, const float* __restrict__ scores,
              const int* __restrict__ params,
              const int* __restrict__ hiBase, const int* __restrict__ eqBase,
              float* __restrict__ sel, float* __restrict__ idxOut) {
    const int tid = threadIdx.x;
    const int blk = blockIdx.x;
    const int b   = blk >> 8;
    const int t   = (blk & 255) * 256 + tid;

    const int cut       = params[b * 4 + 0];
    const int above     = params[b * 4 + 1];
    const int remaining = params[b * 4 + 2];

    const int bin  = score_bin(scores[(size_t)b * SS + t]);
    const int isHi = (bin > cut)  ? 1 : 0;
    const int isEq = (bin == cut) ? 1 : 0;

    // intra-block inclusive scans (Hillis-Steele over 256)
    __shared__ int sh[256], se[256];
    sh[tid] = isHi; se[tid] = isEq;
    __syncthreads();
#pragma unroll
    for (int off = 1; off < 256; off <<= 1) {
        int vh = (tid >= off) ? sh[tid - off] : 0;
        int ve = (tid >= off) ? se[tid - off] : 0;
        __syncthreads();
        sh[tid] += vh; se[tid] += ve;
        __syncthreads();
    }
    const int rankHi = sh[tid] - isHi;
    const int rankEq = se[tid] - isEq;

    int slot = -1;
    if (isHi) {
        slot = hiBase[blk] + rankHi;                    // [0, above)
    } else if (isEq) {
        const int g = eqBase[blk] + rankEq;             // token-order fill of boundary bin
        if (g < remaining) slot = above + g;            // [above, KSEL)
    }

    if (slot >= 0) {
        idxOut[(size_t)b * KSEL + slot] = (float)t;     // exact for t < 2^24
        const float4* src = (const float4*)(emb + ((size_t)b * SS + t) * DD);
        float4*       dst = (float4*)(sel + ((size_t)b * KSEL + slot) * DD);
#pragma unroll
        for (int j = 0; j < DD / 4; ++j) dst[j] = src[j];
    }
}

// ---------------- launch ----------------
extern "C" void kernel_launch(void* const* d_in, const int* in_sizes, int n_in,
                              void* d_out, int out_size, void* d_ws, size_t ws_size,
                              hipStream_t stream) {
    const float* emb = (const float*)d_in[0];
    const float* W1  = (const float*)d_in[1];
    const float* b1  = (const float*)d_in[2];
    const float* W2  = (const float*)d_in[3];
    const float* b2  = (const float*)d_in[4];

    // output layout: selected [BB*KSEL*DD] | indices [BB*KSEL] | scores [BB*SS]
    float* out    = (float*)d_out;
    float* sel    = out;
    float* idxOut = out + (size_t)BB * KSEL * DD;
    float* scores = idxOut + (size_t)BB * KSEL;

    // workspace layout (u32 units)
    unsigned int* hist = (unsigned int*)d_ws;           // BB*NBINS
    int* params = (int*)(hist + (size_t)BB * NBINS);    // BB*4
    int* blkHi  = params + BB * 4;                      // BB*256
    int* blkEq  = blkHi + BB * 256;
    int* hiBase = blkEq + BB * 256;
    int* eqBase = hiBase + BB * 256;

    const int histN = BB * NBINS;
    zero_kernel<<<(histN + 255) / 256, 256, 0, stream>>>(hist, histN);

    score_kernel<<<2048, 256, 0, stream>>>(emb, W1, b1, W2, b2, scores, hist);

    threshold_kernel<<<BB, 32, 0, stream>>>(hist, params);

    count_kernel<<<BB * 256, 256, 0, stream>>>(scores, params, blkHi, blkEq);

    scan_kernel<<<BB, 32, 0, stream>>>(blkHi, blkEq, hiBase, eqBase);

    select_kernel<<<BB * 256, 256, 0, stream>>>(emb, scores, params,
                                                hiBase, eqBase, sel, idxOut);
}
